// EncoderForExport_66864050864555
// MI455X (gfx1250) — compile-verified, matrix operand order checked by reading, not run on hardware
//
#include <hip/hip_runtime.h>

// ---------------------------------------------------------------------------
// CDNA5 (gfx1250) encoder pipeline: conv-as-GEMM + 8 transformer layers +
// down-conv + RVQ. Matrix math via v_wmma_f32_16x16x32_bf16 (wave32); tile
// staging via GLOBAL_LOAD_ASYNC_TO_LDS_B128 (ASYNCcnt) where layout permits.
// ---------------------------------------------------------------------------

typedef __attribute__((ext_vector_type(16))) __bf16 v16bf;
typedef __attribute__((ext_vector_type(8)))  float  v8f;

#define NEGF (-3.4028234663852886e38f)

#define FLAG_BIAS   1
#define FLAG_RESID  2
#define FLAG_GELU   4
#define FLAG_CAUSAL 8
#define FLAG_RVQ    16

// Async DMA: copy 16 bytes per lane, global -> LDS (tracked by ASYNCcnt).
// VDST operand = LDS byte offset (low 32 bits of the generic shared pointer).
__device__ __forceinline__ void async_b128(unsigned lds_off, const float* gptr)
{
    asm volatile("global_load_async_to_lds_b128 %0, %1, off"
                 :: "v"(lds_off), "v"((unsigned long long)(size_t)gptr)
                 : "memory");
}
__device__ __forceinline__ void wait_async0()
{
    asm volatile("s_wait_asynccnt 0" ::: "memory");
}

// ---------------------------------------------------------------------------
// Batched WMMA GEMM: C = epilogue(alpha * A @ B)
//   Block: 256 threads (8 waves). Block tile 128(M) x 64(N), K-step 32.
//   Wave grid 4x2; each wave owns a 32x32 output tile = 4 accumulators,
//   issuing 4 v_wmma_f32_16x16x32_bf16 per K-step.
//   LDS: Asf[128][32] f32, Bsf[64][32] f32, both [row][k] so a fragment is
//   four aligned ds_load_b128 + bf16 packing (lane l: row=l&15, k in
//   {h8+0..7, h8+16..23}, h8 = 8*(l>=16)) per ISA 7.12.2.
//   A tiles (always) and B tiles (OPB==1) stream in via async-to-LDS DMA;
//   edge tiles get a conditional zero-fix pass (async copies cannot mask).
//   OPB: 0 -> B global layout [K][N] (row-major weights; manual transpose)
//        1 -> B global layout [N][K] (conv filters / codebooks / K^T): async
//        2 -> down-conv gather: B[n*1024 + (k&511)*2 + (k>>9)]
//   Requires Kd % 4 == 0 and N % 4 == 0 (true for every call here).
// ---------------------------------------------------------------------------
template <int OPB>
__global__ __launch_bounds__(256) void wmma_gemm_kernel(
    const float* __restrict__ A, const float* __restrict__ Bm,
    float* __restrict__ C,
    const float* __restrict__ bias, const float* __restrict__ resid,
    const float* __restrict__ extra,
    int M, int N, int Kd, int lda, int ldb, int ldc,
    long sAo, long sAi, long sBo, long sBi, long sCo, long sCi,
    int inner, float alpha, int flags)
{
    __shared__ float Asf[128 * 32];
    __shared__ float Bsf[64 * 32];

    const int z  = blockIdx.z;
    const int zo = z / inner, zi = z % inner;
    const float* Ab = A  + (long)zo * sAo + (long)zi * sAi;
    const float* Bb = Bm + (long)zo * sBo + (long)zi * sBi;
    float*       Cb = C  + (long)zo * sCo + (long)zi * sCi;
    const float* Rb = resid ? (resid + (long)zo * sCo + (long)zi * sCi) : (const float*)0;

    const int tid  = threadIdx.x;
    const int wave = tid >> 5;
    const int lane = tid & 31;
    const int half = lane >> 4;
    const int l16  = lane & 15;
    const int h8   = half * 8;

    const int rowBlock = blockIdx.y * 128;
    const int colBlock = blockIdx.x * 64;

    const int wr = (wave >> 1) * 32;   // wave row offset in block tile
    const int wc = (wave & 1) * 32;    // wave col offset in block tile

    // staging coordinates: 32 rows x (8 threads * 16B) per group
    const int sr = tid >> 3;           // 0..31
    const int sk = (tid & 7) << 2;     // 0,4,...,28 (floats)

    v8f acc[2][2] = {};

    for (int k0 = 0; k0 < Kd; k0 += 32) {
        __syncthreads();
        const int gk  = k0 + sk;
        const int okk = (gk < Kd);           // Kd%4==0 -> covers gk+3
        const int gkc = okk ? gk : 0;

        // ---- A tile: 128x32 f32 via async DMA (clamped addresses) ----
#pragma unroll
        for (int g = 0; g < 4; ++g) {
            const int rl   = g * 32 + sr;
            const int grow = rowBlock + rl;
            const float* src = Ab + (long)(grow < M ? grow : 0) * lda + gkc;
            async_b128((unsigned)(size_t)&Asf[rl * 32 + sk], src);
        }
        // ---- B tile: 64x32 f32 into Bsf[col][k] ----
        if (OPB == 1) {
#pragma unroll
            for (int g = 0; g < 2; ++g) {
                const int cl   = g * 32 + sr;
                const int gcol = colBlock + cl;
                const float* src = Bb + (long)(gcol < N ? gcol : 0) * ldb + gkc;
                async_b128((unsigned)(size_t)&Bsf[cl * 32 + sk], src);
            }
        } else if (OPB == 0) {
            // global [k][n]: coalesced along n, transposed scatter into LDS
            const int kk = tid >> 4;             // 0..15
            const int n4 = (tid & 15) << 2;      // 0..60
#pragma unroll
            for (int g = 0; g < 2; ++g) {
                const int kl   = g * 16 + kk;
                const int gk2  = k0 + kl;
                const int gcol = colBlock + n4;
                const int ok   = (gk2 < Kd) && (gcol < N);   // N%4==0 -> +3 safe
                const long idx = (long)(gk2 < Kd ? gk2 : 0) * ldb + (gcol < N ? gcol : 0);
                float4 t = *(const float4*)(Bb + idx);
                __builtin_prefetch(Bb + idx + (long)32 * ldb, 0, 0);
                if (!ok) { t.x = 0.f; t.y = 0.f; t.z = 0.f; t.w = 0.f; }
                Bsf[(n4 + 0) * 32 + kl] = t.x;
                Bsf[(n4 + 1) * 32 + kl] = t.y;
                Bsf[(n4 + 2) * 32 + kl] = t.z;
                Bsf[(n4 + 3) * 32 + kl] = t.w;
            }
        } else { // OPB == 2: down-conv filter gather (stride-2 float elements)
#pragma unroll
            for (int i = 0; i < 8; ++i) {
                const int e    = tid + i * 256;     // 0..2047
                const int cl   = e >> 5;
                const int kl   = e & 31;
                const int gcol = colBlock + cl;
                const int gk2  = k0 + kl;
                const int ok   = (gcol < N) && (gk2 < Kd);
                const long idx = (long)(ok ? gcol : 0) * 1024 +
                                 (ok ? ((gk2 & 511) * 2 + (gk2 >> 9)) : 0);
                const float t = Bb[idx];
                Bsf[cl * 32 + kl] = ok ? t : 0.f;
            }
        }

        wait_async0();   // all this wave's DMA landed in LDS

        // ---- zero-fix OOB slots of async tiles (edge blocks / K tail only) ----
        if (!okk || rowBlock + 128 > M) {
            const float4 zz = {0.f, 0.f, 0.f, 0.f};
#pragma unroll
            for (int g = 0; g < 4; ++g) {
                const int rl = g * 32 + sr;
                if (!okk || (rowBlock + rl) >= M)
                    *(float4*)&Asf[rl * 32 + sk] = zz;
            }
        }
        if (OPB == 1 && (!okk || colBlock + 64 > N)) {
            const float4 zz = {0.f, 0.f, 0.f, 0.f};
#pragma unroll
            for (int g = 0; g < 2; ++g) {
                const int cl = g * 32 + sr;
                if (!okk || (colBlock + cl) >= N)
                    *(float4*)&Bsf[cl * 32 + sk] = zz;
            }
        }
        __syncthreads();

        // ---- fragments: 4 aligned 16B LDS loads + bf16 pack, no bounds logic ----
        auto mkfrag = [](const float* p) -> v16bf {
            const float4 f0 = *(const float4*)(p);
            const float4 f1 = *(const float4*)(p + 4);
            const float4 f2 = *(const float4*)(p + 16);
            const float4 f3 = *(const float4*)(p + 20);
            v16bf r;
            r[0]  = (__bf16)f0.x; r[1]  = (__bf16)f0.y; r[2]  = (__bf16)f0.z; r[3]  = (__bf16)f0.w;
            r[4]  = (__bf16)f1.x; r[5]  = (__bf16)f1.y; r[6]  = (__bf16)f1.z; r[7]  = (__bf16)f1.w;
            r[8]  = (__bf16)f2.x; r[9]  = (__bf16)f2.y; r[10] = (__bf16)f2.z; r[11] = (__bf16)f2.w;
            r[12] = (__bf16)f3.x; r[13] = (__bf16)f3.y; r[14] = (__bf16)f3.z; r[15] = (__bf16)f3.w;
            return r;
        };
        const v16bf a0 = mkfrag(&Asf[(wr + l16) * 32 + h8]);
        const v16bf a1 = mkfrag(&Asf[(wr + 16 + l16) * 32 + h8]);
        const v16bf b0 = mkfrag(&Bsf[(wc + l16) * 32 + h8]);
        const v16bf b1 = mkfrag(&Bsf[(wc + 16 + l16) * 32 + h8]);

        acc[0][0] = __builtin_amdgcn_wmma_f32_16x16x32_bf16(false, a0, false, b0, (short)0, acc[0][0], false, false);
        acc[0][1] = __builtin_amdgcn_wmma_f32_16x16x32_bf16(false, a0, false, b1, (short)0, acc[0][1], false, false);
        acc[1][0] = __builtin_amdgcn_wmma_f32_16x16x32_bf16(false, a1, false, b0, (short)0, acc[1][0], false, false);
        acc[1][1] = __builtin_amdgcn_wmma_f32_16x16x32_bf16(false, a1, false, b1, (short)0, acc[1][1], false, false);
    }

    // ---- epilogue ----
#pragma unroll
    for (int sm = 0; sm < 2; ++sm) {
#pragma unroll
        for (int sn = 0; sn < 2; ++sn) {
            const int col = colBlock + wc + sn * 16 + l16;
            if (col < N) {
#pragma unroll
                for (int r = 0; r < 8; ++r) {
                    const int row = rowBlock + wr + sm * 16 + r + half * 8;
                    if (row < M) {
                        float v = acc[sm][sn][r] * alpha;
                        if (flags & FLAG_RVQ)  v += extra[col];     // + ||codebook_col||^2
                        if (flags & FLAG_BIAS) v += bias[col];
                        if (flags & FLAG_CAUSAL) { if (col > row) v += NEGF; }
                        if (flags & FLAG_GELU) {
                            const float x = v;
                            const float t = tanhf(0.7978845608028654f * (x + 0.044715f * x * x * x));
                            v = 0.5f * x * (1.0f + t);
                        }
                        if (flags & FLAG_RESID) v += Rb[(long)row * ldc + col];
                        Cb[(long)row * ldc + col] = v;
                    }
                }
            }
        }
    }
}

// ---------------------------------------------------------------------------
// LayerNorm over rows of 512, one wave per row (wave32: 16 elems/lane).
// ---------------------------------------------------------------------------
__global__ __launch_bounds__(256) void layernorm_kernel(
    const float* __restrict__ x, const float* __restrict__ w,
    const float* __restrict__ b, float* __restrict__ out, int rows)
{
    const int wave = threadIdx.x >> 5, lane = threadIdx.x & 31;
    const int row = blockIdx.x * 8 + wave;
    if (row >= rows) return;
    const float* xr = x + (long)row * 512;
    float vals[16];
    float s = 0.f;
#pragma unroll
    for (int i = 0; i < 16; ++i) { vals[i] = xr[lane + i * 32]; s += vals[i]; }
#pragma unroll
    for (int m = 16; m >= 1; m >>= 1) s += __shfl_xor(s, m, 32);
    const float mean = s * (1.f / 512.f);
    float v = 0.f;
#pragma unroll
    for (int i = 0; i < 16; ++i) { const float d = vals[i] - mean; v += d * d; }
#pragma unroll
    for (int m = 16; m >= 1; m >>= 1) v += __shfl_xor(v, m, 32);
    const float inv = rsqrtf(v * (1.f / 512.f) + 1e-5f);
    float* orow = out + (long)row * 512;
#pragma unroll
    for (int i = 0; i < 16; ++i) {
        const int c = lane + i * 32;
        orow[c] = (vals[i] - mean) * inv * w[c] + b[c];
    }
}

// ---------------------------------------------------------------------------
// RoPE applied in place to q and k, layout [B,T,H*DH]; DH=64 (32 pairs).
// ---------------------------------------------------------------------------
__global__ __launch_bounds__(256) void rope_kernel(float* __restrict__ q,
                                                   float* __restrict__ k, int total)
{
    const int idx = blockIdx.x * blockDim.x + threadIdx.x;
    if (idx >= total) return;
    const int i   = idx & 31;        // rotary index
    const int tmp = idx >> 5;
    const int hh  = tmp & 7;         // head
    const int bt  = tmp >> 3;        // b*T + t
    const int t   = bt % 500;
    // inv_freq = 10000^(-i/32) = exp(-i * ln(10000)/32)
    const float freq = __expf(-(float)i * 0.28782313662425572f);
    const float ang = (float)t * freq;
    const float c = cosf(ang), s = sinf(ang);
    const long base = (long)bt * 512 + hh * 64 + i;
    const float q1 = q[base], q2 = q[base + 32];
    q[base] = q1 * c - q2 * s;
    q[base + 32] = q2 * c + q1 * s;
    const float k1 = k[base], k2 = k[base + 32];
    k[base] = k1 * c - k2 * s;
    k[base + 32] = k2 * c + k1 * s;
}

// ---------------------------------------------------------------------------
// Row softmax in place; one wave per row of `len`.
// ---------------------------------------------------------------------------
__global__ __launch_bounds__(256) void softmax_kernel(float* __restrict__ s,
                                                      int rows, int len)
{
    const int wave = threadIdx.x >> 5, lane = threadIdx.x & 31;
    const int row = blockIdx.x * 8 + wave;
    if (row >= rows) return;
    float* p = s + (long)row * len;
    float mx = NEGF;
    for (int j = lane; j < len; j += 32) mx = fmaxf(mx, p[j]);
#pragma unroll
    for (int m = 16; m >= 1; m >>= 1) mx = fmaxf(mx, __shfl_xor(mx, m, 32));
    float sum = 0.f;
    for (int j = lane; j < len; j += 32) {
        const float e = __expf(p[j] - mx);
        p[j] = e;
        sum += e;
    }
#pragma unroll
    for (int m = 16; m >= 1; m >>= 1) sum += __shfl_xor(sum, m, 32);
    const float inv = 1.f / sum;
    for (int j = lane; j < len; j += 32) p[j] *= inv;
}

// ---------------------------------------------------------------------------
// Codebook squared norms: rows of 256, one wave per row.
// ---------------------------------------------------------------------------
__global__ __launch_bounds__(256) void sqnorm_kernel(const float* __restrict__ cb,
                                                     float* __restrict__ out, int rows)
{
    const int wave = threadIdx.x >> 5, lane = threadIdx.x & 31;
    const int row = blockIdx.x * 8 + wave;
    if (row >= rows) return;
    const float* p = cb + (long)row * 256;
    float s = 0.f;
#pragma unroll
    for (int i = 0; i < 8; ++i) { const float v = p[lane + i * 32]; s += v * v; }
#pragma unroll
    for (int m = 16; m >= 1; m >>= 1) s += __shfl_xor(s, m, 32);
    if (lane == 0) out[row] = s;
}

// ---------------------------------------------------------------------------
// RVQ argmin over 2048 dists per row + residual update + code emission.
// First-index tie-break to match jnp.argmin. One wave per row.
// ---------------------------------------------------------------------------
__global__ __launch_bounds__(256) void argmin_update_kernel(
    const float* __restrict__ dist, const float* __restrict__ cb,
    float* __restrict__ res, int* __restrict__ codes, int slot, int rows)
{
    const int wave = threadIdx.x >> 5, lane = threadIdx.x & 31;
    const int row = blockIdx.x * 8 + wave;
    if (row >= rows) return;
    const float* dr = dist + (long)row * 2048;
    float best = 3.402823466e38f;
    int bi = 0x7fffffff;
    for (int j = lane; j < 2048; j += 32) {
        const float v = dr[j];
        if (v < best || (v == best && j < bi)) { best = v; bi = j; }
    }
#pragma unroll
    for (int m = 16; m >= 1; m >>= 1) {
        const float ov = __shfl_xor(best, m, 32);
        const int   oi = __shfl_xor(bi, m, 32);
        if (ov < best || (ov == best && oi < bi)) { best = ov; bi = oi; }
    }
    // bi now uniform across the wave
    const float* cr = cb + (long)bi * 256;
    float* rr = res + (long)row * 256;
#pragma unroll
    for (int i = 0; i < 8; ++i) {
        const int c = lane + i * 32;
        rr[c] -= cr[c];
    }
    if (lane == 0) {
        const int b = row / 250, t = row % 250;
        codes[((long)b * 8 + slot) * 250 + t] = bi;
    }
}

// ---------------------------------------------------------------------------
// Host side
// ---------------------------------------------------------------------------
static inline void gemm(hipStream_t st, const float* A, const float* B, float* C,
                        const float* bias, const float* resid, const float* extra,
                        int M, int N, int Kd, int lda, int ldb, int ldc,
                        long sAo, long sAi, long sBo, long sBi, long sCo, long sCi,
                        int outer, int inner, int opB, float alpha, int flags)
{
    dim3 g((N + 63) / 64, (M + 127) / 128, outer * inner);
    if (opB == 0)
        wmma_gemm_kernel<0><<<g, 256, 0, st>>>(A, B, C, bias, resid, extra,
            M, N, Kd, lda, ldb, ldc, sAo, sAi, sBo, sBi, sCo, sCi, inner, alpha, flags);
    else if (opB == 1)
        wmma_gemm_kernel<1><<<g, 256, 0, st>>>(A, B, C, bias, resid, extra,
            M, N, Kd, lda, ldb, ldc, sAo, sAi, sBo, sBi, sCo, sCi, inner, alpha, flags);
    else
        wmma_gemm_kernel<2><<<g, 256, 0, st>>>(A, B, C, bias, resid, extra,
            M, N, Kd, lda, ldb, ldc, sAo, sAi, sBo, sBi, sCo, sCi, inner, alpha, flags);
}

extern "C" void kernel_launch(void* const* d_in, const int* in_sizes, int n_in,
                              void* d_out, int out_size, void* d_ws, size_t ws_size,
                              hipStream_t stream)
{
    (void)in_sizes; (void)n_in; (void)out_size; (void)ws_size;

    const float* audio      = (const float*)d_in[0];
    const float* conv_w     = (const float*)d_in[1];
    const float* conv_b     = (const float*)d_in[2];
    const float* ln1_w      = (const float*)d_in[3];
    const float* ln1_b      = (const float*)d_in[4];
    const float* wq         = (const float*)d_in[5];
    const float* wk         = (const float*)d_in[6];
    const float* wv         = (const float*)d_in[7];
    const float* wo         = (const float*)d_in[8];
    const float* ln2_w      = (const float*)d_in[9];
    const float* ln2_b      = (const float*)d_in[10];
    const float* fc1_w      = (const float*)d_in[11];
    const float* fc1_b      = (const float*)d_in[12];
    const float* fc2_w      = (const float*)d_in[13];
    const float* fc2_b      = (const float*)d_in[14];
    const float* down_w     = (const float*)d_in[15];
    const float* down_b     = (const float*)d_in[16];
    const float* sem_proj_w = (const float*)d_in[17];
    const float* sem_cb     = (const float*)d_in[18];
    const float* ac_proj_w  = (const float*)d_in[19];
    const float* ac_cb      = (const float*)d_in[20];

    int*   codes = (int*)d_out;     // [B=8][8][250] int32 codes
    float* ws    = (float*)d_ws;

    // dims
    const int B = 8, T = 500, D = 512, H = 8, FF = 2048;
    const int M = B * T;             // 4000
    const long TD = (long)T * D;     // 256000
    const int T2 = 250, M2 = B * T2; // 2000
    const int CB = 2048, CD = 256;

    // workspace layout (floats); scores / ffbuf / dist share one region
    float* h      = ws;                          // 2,048,000
    float* hn     = h  + (long)M * D;            // 2,048,000
    float* q      = hn + (long)M * D;            // 2,048,000
    float* k      = q  + (long)M * D;            // 2,048,000
    float* v      = k  + (long)M * D;            // 2,048,000
    float* o      = v  + (long)M * D;            // 2,048,000
    float* big    = o  + (long)M * D;            // 16,000,000 (scores|ff|dist)
    float* scores = big;
    float* ffbuf  = big;
    float* dist   = big;
    float* emb    = big + 16000000;              // 1,024,000
    float* res    = emb + (long)M2 * D;          //   512,000
    float* cbn    = res + (long)M2 * CD;         //    16,384 (8 stages x 2048)

    const long Z = 0;

    // 1) Stride-480 / K=480 conv == GEMM: audio [4000x480] @ conv_w^T -> h [B,T,D]
    gemm(stream, audio, conv_w, h, conv_b, nullptr, nullptr,
         M, D, 480, 480, 480, D, Z, Z, Z, Z, Z, Z, 1, 1, 1, 1.f, FLAG_BIAS);

    // 2) Transformer layers
    for (int i = 0; i < 8; ++i) {
        layernorm_kernel<<<(M + 7) / 8, 256, 0, stream>>>(h, ln1_w + i * D, ln1_b + i * D, hn, M);

        gemm(stream, hn, wq + (long)i * D * D, q, nullptr, nullptr, nullptr,
             M, D, D, D, D, D, Z, Z, Z, Z, Z, Z, 1, 1, 0, 1.f, 0);
        gemm(stream, hn, wk + (long)i * D * D, k, nullptr, nullptr, nullptr,
             M, D, D, D, D, D, Z, Z, Z, Z, Z, Z, 1, 1, 0, 1.f, 0);
        gemm(stream, hn, wv + (long)i * D * D, v, nullptr, nullptr, nullptr,
             M, D, D, D, D, D, Z, Z, Z, Z, Z, Z, 1, 1, 0, 1.f, 0);

        rope_kernel<<<(1024000 + 255) / 256, 256, 0, stream>>>(q, k, 1024000);

        // scores[b,h,q,k] = (Q.K^T)/8 + causal
        gemm(stream, q, k, scores, nullptr, nullptr, nullptr,
             T, T, 64, D, D, T,
             TD, 64, TD, 64, (long)H * T * T, (long)T * T,
             B, H, 1, 0.125f, FLAG_CAUSAL);

        softmax_kernel<<<(B * H * T + 7) / 8, 256, 0, stream>>>(scores, B * H * T, T);

        // o = P @ V
        gemm(stream, scores, v, o, nullptr, nullptr, nullptr,
             T, 64, T, T, D, D,
             (long)H * T * T, (long)T * T, TD, 64, TD, 64,
             B, H, 0, 1.f, 0);

        // h += o @ Wo
        gemm(stream, o, wo + (long)i * D * D, h, nullptr, h, nullptr,
             M, D, D, D, D, D, Z, Z, Z, Z, Z, Z, 1, 1, 0, 1.f, FLAG_RESID);

        layernorm_kernel<<<(M + 7) / 8, 256, 0, stream>>>(h, ln2_w + i * D, ln2_b + i * D, hn, M);

        // ff = gelu(hn @ fc1 + b1)
        gemm(stream, hn, fc1_w + (long)i * D * FF, ffbuf, fc1_b + (long)i * FF, nullptr, nullptr,
             M, FF, D, D, FF, FF, Z, Z, Z, Z, Z, Z, 1, 1, 0, 1.f, FLAG_BIAS | FLAG_GELU);

        // h += ff @ fc2 + b2
        gemm(stream, ffbuf, fc2_w + (long)i * FF * D, h, fc2_b + (long)i * D, h, nullptr,
             M, D, FF, FF, D, D, Z, Z, Z, Z, Z, Z, 1, 1, 0, 1.f, FLAG_BIAS | FLAG_RESID);
    }

    // 3) Down-conv (stride 2, k=2) == GEMM: h [2000x1024] @ gather(down_w) -> emb [B,T2,D]
    gemm(stream, h, down_w, emb, down_b, nullptr, nullptr,
         M2, D, 1024, 1024, 1024, D, Z, Z, Z, Z, Z, Z, 1, 1, 2, 1.f, FLAG_BIAS);

    // 4) Codebook norms: sem stage 0 + ac stages 0..6 (only first 8 codes kept)
    sqnorm_kernel<<<(CB + 7) / 8, 256, 0, stream>>>(sem_cb, cbn, CB);
    sqnorm_kernel<<<(7 * CB + 7) / 8, 256, 0, stream>>>(ac_cb, cbn + CB, 7 * CB);

    // 5) Semantic RVQ (1 stage)
    gemm(stream, emb, sem_proj_w, res, nullptr, nullptr, nullptr,
         M2, CD, D, D, D, CD, Z, Z, Z, Z, Z, Z, 1, 1, 1, 1.f, 0);
    gemm(stream, res, sem_cb, dist, nullptr, nullptr, cbn,
         M2, CB, CD, CD, CD, CB, Z, Z, Z, Z, Z, Z, 1, 1, 1, -2.f, FLAG_RVQ);
    argmin_update_kernel<<<(M2 + 7) / 8, 256, 0, stream>>>(dist, sem_cb, res, codes, 0, M2);

    // 6) Acoustic RVQ (stages 0..6)
    gemm(stream, emb, ac_proj_w, res, nullptr, nullptr, nullptr,
         M2, CD, D, D, D, CD, Z, Z, Z, Z, Z, Z, 1, 1, 1, 1.f, 0);
    for (int j = 0; j < 7; ++j) {
        const float* cbj = ac_cb + (long)j * CB * CD;
        gemm(stream, res, cbj, dist, nullptr, nullptr, cbn + CB + (long)j * CB,
             M2, CB, CD, CD, CD, CB, Z, Z, Z, Z, Z, Z, 1, 1, 1, -2.f, FLAG_RVQ);
        argmin_update_kernel<<<(M2 + 7) / 8, 256, 0, stream>>>(dist, cbj, res, codes, 1 + j, M2);
    }
}
